// LSTMOnlineModel_54073638257123
// MI455X (gfx1250) — matrix operand order repaired and checked by reading, not compiled, not run on hardware
//
#include <hip/hip_runtime.h>
#include <hip/hip_bf16.h>

// Problem constants (from the reference)
#define BATCH    4096
#define TSTEPS   512
#define IN_D     7
#define HID      64
#define GATES    (4 * HID)   // 256
#define OUTD     4
#define ROWS     16          // batch rows per block (one WMMA M-tile)
#define NTHREADS 128         // 4 waves of 32; wave w owns gate w (i,f,g,o)
#define HPAD     72          // padded bf16 row stride for h staging (144B rows, 16B aligned)

typedef __bf16 bf16;
typedef __attribute__((ext_vector_type(16))) __bf16 v16bf;
typedef __attribute__((ext_vector_type(8)))  __bf16 v8bf;
typedef __attribute__((ext_vector_type(8)))  float  v8f;

static __device__ __forceinline__ v8f wmma_bf16(v16bf a, v16bf b, v8f c) {
  // D = A(16x32 bf16) * B(32x16 bf16) + C(16x16 f32)
  return __builtin_amdgcn_wmma_f32_16x16x32_bf16(false, a, false, b, (short)0, c, false, false);
}

static __device__ __forceinline__ float fast_sigmoid(float x) {
  return __builtin_amdgcn_rcpf(1.0f + __expf(-x));
}
static __device__ __forceinline__ float fast_tanh(float x) {
  return 2.0f * __builtin_amdgcn_rcpf(1.0f + __expf(-2.0f * x)) - 1.0f;
}

// ---- B fragment (32x16) of B = W^T for output cols [n0,n0+16), k in [k0,k0+32).
// W is [N x Kdim] row-major f32 in global memory; kvalid zero-pads short K (W_ih0, K=7).
// Contiguous-K order used here (same as A): lane l -> col n = n0 + (l&15); lanes 0-15 hold K=k0..k0+15 (2 per VGPR),
// lanes 16-31 hold K=k0+16..k0+31. Each lane = 16 contiguous elements of W row n.
static __device__ __forceinline__ v16bf load_Bfrag(const float* __restrict__ W, int Kdim,
                                                   int kvalid, int n0, int k0, int lane) {
  int n  = n0 + (lane & 15);
  int kh = (lane >> 4) << 4;   // 0 or 16
  v16bf b;
#pragma unroll
  for (int j = 0; j < 16; ++j) {
    int k = k0 + kh + j;
    float w = (k < kvalid) ? W[n * Kdim + k] : 0.0f;
    b[j] = (bf16)w;
  }
  return b;
}

// ---- A fragment (16x32) starting at column k0 of a row-major bf16 LDS array (stride HPAD).
// Contiguous-K A order used here: lane l -> row l&15; lanes 0-15: K = k0..k0+15,
// lanes 16-31: K = k0+16..k0+31 (matches B). Two aligned 16-byte LDS loads.
static __device__ __forceinline__ v16bf load_Afrag(const bf16* base, int k0, int lane) {
  int row = lane & 15;
  int kb  = (lane >> 4) << 4;  // 0 or 16
  const bf16* p = base + row * HPAD + k0 + kb;
  v8bf lo = *(const v8bf*)p;
  v8bf hi = *(const v8bf*)(p + 8);
  return __builtin_shufflevector(lo, hi, 0,1,2,3,4,5,6,7,8,9,10,11,12,13,14,15);
}

// ---- D (16x16 f32) scatter into the LDS gate buffer.
// Layout: lane l -> col n = n0 + (l&15); VGPR r -> row m = r + 8*(l>=16).
static __device__ __forceinline__ void store_gates(float (*g)[GATES], v8f acc, int n0, int lane) {
  int n  = n0 + (lane & 15);
  int m0 = (lane >> 4) << 3;
#pragma unroll
  for (int r = 0; r < 8; ++r) g[m0 + r][n] = acc[r];
}

__global__ __launch_bounds__(NTHREADS, 1)
void lstm2_fused_kernel(const float* __restrict__ x,
                        const float* __restrict__ W_ih0, const float* __restrict__ W_hh0,
                        const float* __restrict__ b_ih0, const float* __restrict__ b_hh0,
                        const float* __restrict__ W_ih1, const float* __restrict__ W_hh1,
                        const float* __restrict__ b_ih1, const float* __restrict__ b_hh1,
                        const float* __restrict__ W_fc,  const float* __restrict__ b_fc,
                        float* __restrict__ out) {
  __shared__ float sGates[ROWS][GATES];                 // 16 KB, reused by both layers
  __shared__ __align__(16) bf16 sH0[ROWS][HPAD];        // layer0 hidden state (bf16, A layout src)
  __shared__ __align__(16) bf16 sH1[ROWS][HPAD];        // layer1 hidden state
  __shared__ float sH1f[ROWS][HID];                     // f32 copy of last-step h1 for the FC
  __shared__ float sB0[GATES];                          // b_ih0 + b_hh0
  __shared__ float sB1[GATES];                          // b_ih1 + b_hh1

  const int tid  = threadIdx.x;
  const int lane = tid & 31;
  const int wave = tid >> 5;            // gate index: 0=i, 1=f, 2=g, 3=o
  const int b0   = blockIdx.x * ROWS;   // batch-row base of this tile

  // ---------- one-time init ----------
  for (int idx = tid; idx < GATES; idx += NTHREADS) {
    sB0[idx] = b_ih0[idx] + b_hh0[idx];
    sB1[idx] = b_ih1[idx] + b_hh1[idx];
  }
  for (int idx = tid; idx < ROWS * HPAD; idx += NTHREADS) {
    (&sH0[0][0])[idx] = (bf16)0.0f;
    (&sH1[0][0])[idx] = (bf16)0.0f;
  }

  // Register-resident weight B-fragments: each wave holds only its gate's 64 columns.
  // 28 fragments * 8 VGPRs = 224 VGPRs of weights; zero weight traffic in the time loop.
  v16bf Bih0[4];        // K padded 7 -> 32 (one K-step)
  v16bf Bhh0[4][2];     // K = 64 -> two K-steps
  v16bf Bih1[4][2];
  v16bf Bhh1[4][2];
#pragma unroll
  for (int nt = 0; nt < 4; ++nt) {
    int n0 = wave * 64 + nt * 16;
    Bih0[nt]    = load_Bfrag(W_ih0, IN_D, IN_D, n0, 0,  lane);
    Bhh0[nt][0] = load_Bfrag(W_hh0, HID,  HID,  n0, 0,  lane);
    Bhh0[nt][1] = load_Bfrag(W_hh0, HID,  HID,  n0, 32, lane);
    Bih1[nt][0] = load_Bfrag(W_ih1, HID,  HID,  n0, 0,  lane);
    Bih1[nt][1] = load_Bfrag(W_ih1, HID,  HID,  n0, 32, lane);
    Bhh1[nt][0] = load_Bfrag(W_hh1, HID,  HID,  n0, 0,  lane);
    Bhh1[nt][1] = load_Bfrag(W_hh1, HID,  HID,  n0, 32, lane);
  }

  // Per-thread cell state: thread t owns cells [t*8, t*8+8) of the 16x64 (m,u) grid.
  float c0r[8], c1r[8];
#pragma unroll
  for (int e = 0; e < 8; ++e) { c0r[e] = 0.0f; c1r[e] = 0.0f; }

  __syncthreads();

#pragma unroll 1
  for (int t = 0; t < TSTEPS; ++t) {
    // ---- P1: layer0 gates = x_t @ W_ih0^T + h0 @ W_hh0^T ----
    v16bf xA;
#pragma unroll
    for (int j = 0; j < 16; ++j) xA[j] = (bf16)0.0f;
    if (lane < 16) {
      // lane l = batch row l; K-cols 0..6 of the padded 16x32 A tile (rest zero)
      const float* px = x + ((size_t)(b0 + lane) * TSTEPS + t) * IN_D;
#pragma unroll
      for (int j = 0; j < IN_D; ++j) xA[j] = (bf16)px[j];
      __builtin_prefetch((const void*)(px + 8 * IN_D), 0, 0);  // stream ahead 8 timesteps
    }
    v16bf h0A0 = load_Afrag(&sH0[0][0], 0,  lane);
    v16bf h0A1 = load_Afrag(&sH0[0][0], 32, lane);
#pragma unroll
    for (int nt = 0; nt < 4; ++nt) {
      v8f acc = {0.f, 0.f, 0.f, 0.f, 0.f, 0.f, 0.f, 0.f};
      acc = wmma_bf16(xA,   Bih0[nt],    acc);
      acc = wmma_bf16(h0A0, Bhh0[nt][0], acc);
      acc = wmma_bf16(h0A1, Bhh0[nt][1], acc);
      store_gates(sGates, acc, wave * 64 + nt * 16, lane);
    }
    __syncthreads();

    // ---- P2: layer0 pointwise (i,f,g,o) -> c0, h0 ----
#pragma unroll
    for (int e = 0; e < 8; ++e) {
      int cell = tid * 8 + e;
      int m = cell >> 6, u = cell & 63;
      float ig = fast_sigmoid(sGates[m][u]       + sB0[u]);
      float fg = fast_sigmoid(sGates[m][64 + u]  + sB0[64 + u]);
      float gg = fast_tanh   (sGates[m][128 + u] + sB0[128 + u]);
      float og = fast_sigmoid(sGates[m][192 + u] + sB0[192 + u]);
      float c  = fg * c0r[e] + ig * gg;
      c0r[e]   = c;
      float h  = og * fast_tanh(c);
      sH0[m][u] = (bf16)h;
    }
    __syncthreads();

    // ---- P3: layer1 gates = h0 @ W_ih1^T + h1 @ W_hh1^T ----
    v16bf g0A0 = load_Afrag(&sH0[0][0], 0,  lane);
    v16bf g0A1 = load_Afrag(&sH0[0][0], 32, lane);
    v16bf h1A0 = load_Afrag(&sH1[0][0], 0,  lane);
    v16bf h1A1 = load_Afrag(&sH1[0][0], 32, lane);
#pragma unroll
    for (int nt = 0; nt < 4; ++nt) {
      v8f acc = {0.f, 0.f, 0.f, 0.f, 0.f, 0.f, 0.f, 0.f};
      acc = wmma_bf16(g0A0, Bih1[nt][0], acc);
      acc = wmma_bf16(g0A1, Bih1[nt][1], acc);
      acc = wmma_bf16(h1A0, Bhh1[nt][0], acc);
      acc = wmma_bf16(h1A1, Bhh1[nt][1], acc);
      store_gates(sGates, acc, wave * 64 + nt * 16, lane);
    }
    __syncthreads();

    // ---- P4: layer1 pointwise -> c1, h1 ----
#pragma unroll
    for (int e = 0; e < 8; ++e) {
      int cell = tid * 8 + e;
      int m = cell >> 6, u = cell & 63;
      float ig = fast_sigmoid(sGates[m][u]       + sB1[u]);
      float fg = fast_sigmoid(sGates[m][64 + u]  + sB1[64 + u]);
      float gg = fast_tanh   (sGates[m][128 + u] + sB1[128 + u]);
      float og = fast_sigmoid(sGates[m][192 + u] + sB1[192 + u]);
      float c  = fg * c1r[e] + ig * gg;
      c1r[e]   = c;
      float h  = og * fast_tanh(c);
      sH1[m][u] = (bf16)h;
      if (t == TSTEPS - 1) sH1f[m][u] = h;   // keep f32 last-step h1 for the FC head
    }
    __syncthreads();
  }

  // ---- FC head: out[b0+m][o] = h1_last[m] . W_fc[o] + b_fc[o]  (tiny: 64x4 per tile) ----
  if (tid < ROWS * OUTD) {
    int m = tid >> 2, o = tid & 3;
    float acc = b_fc[o];
#pragma unroll
    for (int u = 0; u < HID; ++u) acc += sH1f[m][u] * W_fc[o * HID + u];
    out[(size_t)(b0 + m) * OUTD + o] = acc;
  }
}

extern "C" void kernel_launch(void* const* d_in, const int* in_sizes, int n_in,
                              void* d_out, int out_size, void* d_ws, size_t ws_size,
                              hipStream_t stream) {
  (void)in_sizes; (void)n_in; (void)out_size; (void)d_ws; (void)ws_size;
  const float* x     = (const float*)d_in[0];
  const float* W_ih0 = (const float*)d_in[1];
  const float* W_hh0 = (const float*)d_in[2];
  const float* b_ih0 = (const float*)d_in[3];
  const float* b_hh0 = (const float*)d_in[4];
  const float* W_ih1 = (const float*)d_in[5];
  const float* W_hh1 = (const float*)d_in[6];
  const float* b_ih1 = (const float*)d_in[7];
  const float* b_hh1 = (const float*)d_in[8];
  const float* W_fc  = (const float*)d_in[9];
  const float* b_fc  = (const float*)d_in[10];
  float* out = (float*)d_out;

  dim3 grid(BATCH / ROWS);   // 256 independent 16-row recurrence chains
  dim3 block(NTHREADS);      // 4 waves: one per gate
  hipLaunchKernelGGL(lstm2_fused_kernel, grid, block, 0, stream,
                     x, W_ih0, W_hh0, b_ih0, b_hh0,
                     W_ih1, W_hh1, b_ih1, b_hh1, W_fc, b_fc, out);
}
